// vision_mamba_61400852464204
// MI455X (gfx1250) — matrix operand order repaired and checked by reading, not compiled, not run on hardware
//
#include <hip/hip_runtime.h>
#include <math.h>

typedef __attribute__((ext_vector_type(16))) _Float16 v16h;
typedef __attribute__((ext_vector_type(8)))  float    v8f;

namespace {

constexpr int Bsz = 8, Cc = 96, Dd = 192, Nn = 16, Rr = 6, Kk = 4;
constexpr int DEPTH = 12, NCLS = 43;
constexpr int Hh = 14, Ww = 14, Ll = 196, ROWS = Bsz * Ll;   // 1568
constexpr int BCW = 32;   // B|C columns fed to the scan (2*N)

// ---------------------------------------------------------------------------
// Fragment helpers (k order used by this code: contiguous k = 16*half + s, wave32):
//   A (16x32 f16): lanes 0-15 row M=lane, halves 0-15 -> K 0-15
//                  lanes 16-31 row M=lane-16, halves -> K 16-31
//   B (32x16 f16): lanes 0-15 N=lane, halves -> K 0-15; lanes 16-31 K 16-31
//   C (16x16 f32): vgpr r: M=r (lanes 0-15) / r+8 (lanes 16-31), N=lane&15
// ---------------------------------------------------------------------------
__device__ __forceinline__ v16h zero16() {
  v16h a;
#pragma unroll
  for (int i = 0; i < 16; ++i) a[i] = (_Float16)0.f;
  return a;
}

// p points at (row base + kc + kAoff); pulls floats [0..15]
// as four b128 loads and packs to the A fragment's 16 halves.
__device__ __forceinline__ v16h load_a16(const float* __restrict__ p) {
  const float4* q = (const float4*)p;
  float4 x0 = q[0], x1 = q[1], x2 = q[2], x3 = q[3];
  v16h a;
  a[0]  = (_Float16)x0.x; a[1]  = (_Float16)x0.y;
  a[2]  = (_Float16)x0.z; a[3]  = (_Float16)x0.w;
  a[4]  = (_Float16)x1.x; a[5]  = (_Float16)x1.y;
  a[6]  = (_Float16)x1.z; a[7]  = (_Float16)x1.w;
  a[8]  = (_Float16)x2.x; a[9]  = (_Float16)x2.y;
  a[10] = (_Float16)x2.z; a[11] = (_Float16)x2.w;
  a[12] = (_Float16)x3.x; a[13] = (_Float16)x3.y;
  a[14] = (_Float16)x3.z; a[15] = (_Float16)x3.w;
  return a;
}

#define WMMA32(a, b, c) \
  __builtin_amdgcn_wmma_f32_16x16x32_f16(false, (a), false, (b), (short)0, (c), false, false)

// Epilogue modes
//   0: C = acc            1: C = softplus(acc + bias[n])  (per-z bias)
//   2: C += acc           3: C = acc + bias[n]
__device__ __forceinline__ void store_c(float* __restrict__ Cm, long idx, float v,
                                        int mode, float bias) {
  if (mode == 0) {
    Cm[idx] = v;
  } else if (mode == 1) {
    float s = v + bias;
    Cm[idx] = (s > 20.f) ? s : log1pf(__expf(s));
  } else if (mode == 2) {
    Cm[idx] += v;
  } else {
    Cm[idx] = v + bias;
  }
}

// ---------------------------------------------------------------------------
// GEMM: C[z] (MxN, f32) = A[z] (MxK, f32) * Bw[z] (f16, packed [n][K]).
// One wave computes a 32x32 tile: 2 A-frags x 2 B-frags -> 4 WMMAs per chunk.
// Fast path is software-pipelined: chunk k+1 fragment loads are issued before
// chunk k's WMMAs so VMEM overlaps the matrix pipe. Kt must be % 32.
// ---------------------------------------------------------------------------
__global__ void gemm_wmma(const float* __restrict__ A, long sAz,
                          const _Float16* __restrict__ Bw, long sBz,
                          float* __restrict__ Cm, long sCz,
                          int M, int N, int Kt, int lda, int ldc,
                          int mode, const float* __restrict__ bias, long sBiasZ)
{
  const int lane = (int)(threadIdx.x & 31u);
  const int row0 = blockIdx.x * 32;
  const int col0 = blockIdx.y * 32;
  const int z = blockIdx.z;
  A  += (long)z * sAz;
  Bw += (long)z * sBz;
  Cm += (long)z * sCz;
  if (bias) bias += (long)z * sBiasZ;

  const int half16 = lane >> 4;        // 0 or 1
  const int l15 = lane & 15;
  const int kAoff = half16 * 16;
  const int kBoff = half16 * 16;
  const int mA0 = row0 + l15, mA1 = mA0 + 16;
  const int nB0 = col0 + l15, nB1 = nB0 + 16;

  v8f c00 = {0.f,0.f,0.f,0.f,0.f,0.f,0.f,0.f};
  v8f c01 = c00, c10 = c00, c11 = c00;

  const bool full = (row0 + 32 <= M) && (col0 + 32 <= N);
  if (full) {
    const float*    pa0 = A  + (long)mA0 * lda + kAoff;
    const float*    pa1 = A  + (long)mA1 * lda + kAoff;
    const _Float16* pb0 = Bw + (long)nB0 * Kt  + kBoff;
    const _Float16* pb1 = Bw + (long)nB1 * Kt  + kBoff;
    v16h a0 = load_a16(pa0);
    v16h a1 = load_a16(pa1);
    v16h b0 = *(const v16h*)(pb0);
    v16h b1 = *(const v16h*)(pb1);
    for (int kc = 32; kc < Kt; kc += 32) {
      v16h na0 = load_a16(pa0 + kc);          // prefetch next chunk
      v16h na1 = load_a16(pa1 + kc);
      v16h nb0 = *(const v16h*)(pb0 + kc);
      v16h nb1 = *(const v16h*)(pb1 + kc);
      c00 = WMMA32(a0, b0, c00);
      c01 = WMMA32(a0, b1, c01);
      c10 = WMMA32(a1, b0, c10);
      c11 = WMMA32(a1, b1, c11);
      a0 = na0; a1 = na1; b0 = nb0; b1 = nb1;
    }
    c00 = WMMA32(a0, b0, c00);
    c01 = WMMA32(a0, b1, c01);
    c10 = WMMA32(a1, b0, c10);
    c11 = WMMA32(a1, b1, c11);
  } else {
    for (int kc = 0; kc < Kt; kc += 32) {
      v16h a0 = (mA0 < M) ? load_a16(A + (long)mA0 * lda + kc + kAoff) : zero16();
      v16h a1 = (mA1 < M) ? load_a16(A + (long)mA1 * lda + kc + kAoff) : zero16();
      v16h b0 = (nB0 < N) ? *(const v16h*)(Bw + (long)nB0 * Kt + kc + kBoff) : zero16();
      v16h b1 = (nB1 < N) ? *(const v16h*)(Bw + (long)nB1 * Kt + kc + kBoff) : zero16();
      c00 = WMMA32(a0, b0, c00);
      c01 = WMMA32(a0, b1, c01);
      c10 = WMMA32(a1, b0, c10);
      c11 = WMMA32(a1, b1, c11);
    }
  }

  const float bias0 = (bias && nB0 < N) ? bias[nB0] : 0.f;
  const float bias1 = (bias && nB1 < N) ? bias[nB1] : 0.f;
  const int mb0 = row0 + half16 * 8;
  const int mb1 = mb0 + 16;
#pragma unroll
  for (int r = 0; r < 8; ++r) {
    const int m0 = mb0 + r, m1 = mb1 + r;
    if (nB0 < N) {
      if (m0 < M) store_c(Cm, (long)m0 * ldc + nB0, c00[r], mode, bias0);
      if (m1 < M) store_c(Cm, (long)m1 * ldc + nB0, c10[r], mode, bias0);
    }
    if (nB1 < N) {
      if (m0 < M) store_c(Cm, (long)m0 * ldc + nB1, c01[r], mode, bias1);
      if (m1 < M) store_c(Cm, (long)m1 * ldc + nB1, c11[r], mode, bias1);
    }
  }
}

// ------------------------- weight prep (f32 -> f16 packed [n][K]) ----------
__global__ void prep_patch(const float* __restrict__ w, _Float16* __restrict__ o) {
  int idx = blockIdx.x * blockDim.x + threadIdx.x;
  if (idx < Cc * 768) o[idx] = (_Float16)w[idx];           // already [n][k]
}
__global__ void prep_inproj(const float* __restrict__ w, _Float16* __restrict__ o) {
  int idx = blockIdx.x * blockDim.x + threadIdx.x;         // 12*384*96
  if (idx >= DEPTH * 2 * Dd * Cc) return;
  int c = idx % Cc; int r = idx / Cc;
  int n = r % (2 * Dd); int i = r / (2 * Dd);
  o[idx] = (_Float16)w[((long)i * Cc + c) * (2 * Dd) + n];
}
__global__ void prep_outproj(const float* __restrict__ w, _Float16* __restrict__ o) {
  int idx = blockIdx.x * blockDim.x + threadIdx.x;         // 12*96*192
  if (idx >= DEPTH * Cc * Dd) return;
  int k = idx % Dd; int r = idx / Dd;
  int n = r % Cc; int i = r / Cc;
  o[idx] = (_Float16)w[((long)i * Dd + k) * Cc + n];
}
__global__ void prep_bc(const float* __restrict__ xpw, _Float16* __restrict__ o) {
  int idx = blockIdx.x * blockDim.x + threadIdx.x;         // 48*32*192
  if (idx >= DEPTH * Kk * BCW * Dd) return;
  int dd = idx % Dd; int r = idx / Dd;
  int c = r % BCW; int g = r / BCW;                        // g = i*4 + k
  o[idx] = (_Float16)xpw[((long)g * 38 + Rr + c) * Dd + dd];
}
// fused dt weight: W2[g][d][dd] = sum_r dtw[g,d,r] * xpw[g,r,dd]  (dt_proj o x_proj)
__global__ void prep_dtfused(const float* __restrict__ xpw, const float* __restrict__ dtw,
                             _Float16* __restrict__ o) {
  int idx = blockIdx.x * blockDim.x + threadIdx.x;         // 48*192*192
  if (idx >= DEPTH * Kk * Dd * Dd) return;
  int dd = idx % Dd; int r2 = idx / Dd;
  int d = r2 % Dd; int g = r2 / Dd;
  const float* dw = dtw + ((long)g * Dd + d) * Rr;
  const float* xp = xpw + (long)g * 38 * Dd + dd;
  float s = 0.f;
#pragma unroll
  for (int r = 0; r < Rr; ++r) s += dw[r] * xp[(long)r * Dd];
  o[idx] = (_Float16)s;
}
__global__ void prep_head(const float* __restrict__ w, _Float16* __restrict__ o) {
  int idx = blockIdx.x * blockDim.x + threadIdx.x;         // 43*96
  if (idx >= NCLS * Cc) return;
  int k = idx % Cc; int n = idx / Cc;
  o[idx] = (_Float16)w[k * NCLS + n];
}

// ------------------------- elementwise / special kernels -------------------
__global__ void im2col_patch(const float* __restrict__ x, float* __restrict__ col) {
  int idx = blockIdx.x * blockDim.x + threadIdx.x;
  if (idx >= ROWS * 768) return;
  int m = idx / 768, k = idx - m * 768;
  int b = m / Ll, l = m - b * Ll;
  int ph = l / Ww, pw = l - ph * Ww;
  int ci = k / 256, r = k - ci * 256;
  int py = r / 16, px = r - py * 16;
  col[idx] = x[(((long)b * 3 + ci) * 224 + (ph * 16 + py)) * 224 + (pw * 16 + px)];
}

__global__ void patch_epilogue(float* __restrict__ t, const float* __restrict__ pb,
                               const float* __restrict__ pos) {
  int idx = blockIdx.x * blockDim.x + threadIdx.x;
  if (idx >= ROWS * Cc) return;
  int c = idx % Cc;
  int l = (idx / Cc) % Ll;
  t[idx] += pb[c] + pos[l * Cc + c];
}

__global__ void layernorm_rows(const float* __restrict__ in, float* __restrict__ out,
                               const float* __restrict__ w, const float* __restrict__ b,
                               int rows, int width) {
  int r = blockIdx.x * blockDim.x + threadIdx.x;
  if (r >= rows) return;
  const float* p = in + (long)r * width;
  float mu = 0.f;
  for (int i = 0; i < width; ++i) mu += p[i];
  mu /= (float)width;
  float var = 0.f;
  for (int i = 0; i < width; ++i) { float d = p[i] - mu; var += d * d; }
  var /= (float)width;
  float rs = rsqrtf(var + 1e-6f);
  float* q = out + (long)r * width;
  for (int i = 0; i < width; ++i) q[i] = (p[i] - mu) * rs * w[i] + b[i];
}

// depthwise 3x3 SAME conv + bias + SiLU; writes the 4 direction-ordered copies
__global__ void dwconv_scan_prep(const float* __restrict__ xz, const float* __restrict__ cw,
                                 const float* __restrict__ cb, float* __restrict__ xst) {
  int idx = blockIdx.x * blockDim.x + threadIdx.x;
  if (idx >= ROWS * Dd) return;
  int d = idx % Dd;
  int m = idx / Dd;
  int b = m / Ll, l = m - b * Ll;
  int h = l / Ww, w = l - h * Ww;
  float s = 0.f;
#pragma unroll
  for (int dy = -1; dy <= 1; ++dy) {
    int hh = h + dy;
    if (hh < 0 || hh >= Hh) continue;
#pragma unroll
    for (int dx = -1; dx <= 1; ++dx) {
      int ww = w + dx;
      if (ww < 0 || ww >= Ww) continue;
      s += cw[d * 9 + (dy + 1) * 3 + (dx + 1)] *
           xz[((long)b * Ll + hh * Ww + ww) * (2 * Dd) + d];
    }
  }
  s += cb[d];
  s = s / (1.f + __expf(-s));  // SiLU
  const long kS = (long)ROWS * Dd;
  int l2 = w * Hh + h;
  xst[((long)b * Ll + l) * Dd + d]                      = s;
  xst[kS     + ((long)b * Ll + l2) * Dd + d]            = s;
  xst[2 * kS + ((long)b * Ll + (Ll - 1 - l)) * Dd + d]  = s;
  xst[3 * kS + ((long)b * Ll + (Ll - 1 - l2)) * Dd + d] = s;
}

// Selective scan: one thread per (k,b,d); writes cross-merge-mapped outputs.
__global__ void selective_scan(const float* __restrict__ xst, const float* __restrict__ dts,
                               const float* __restrict__ bc, const float* __restrict__ Alog,
                               const float* __restrict__ Dsk, float* __restrict__ ys) {
  int idx = blockIdx.x * blockDim.x + threadIdx.x;
  if (idx >= Kk * Bsz * Dd) return;  // 6144
  int d = idx % Dd;
  int t = idx / Dd;
  int b = t % Bsz;
  int k = t / Bsz;
  const long kS = (long)ROWS * Dd;
  const float* xp  = xst + (long)k * kS + ((long)b * Ll) * Dd + d;
  const float* dtp = dts + (long)k * kS + ((long)b * Ll) * Dd + d;
  const float* blp = bc + ((long)k * ROWS + (long)b * Ll) * BCW;
  float Av[Nn];
#pragma unroll
  for (int n = 0; n < Nn; ++n) Av[n] = -__expf(Alog[((long)k * Dd + d) * Nn + n]);
  const float Dv = Dsk[k * Dd + d];
  float h[Nn];
#pragma unroll
  for (int n = 0; n < Nn; ++n) h[n] = 0.f;
  float* yk = ys + (long)k * kS;
  for (int l = 0; l < Ll; ++l) {
    const float xv = xp[(long)l * Dd];
    const float dt = dtp[(long)l * Dd];
    const float* bl = blp + (long)l * BCW;
    const float dx = dt * xv;
    float y = 0.f;
#pragma unroll
    for (int n = 0; n < Nn; ++n) {
      h[n] = h[n] * __expf(dt * Av[n]) + dx * bl[n];
      y += h[n] * bl[Nn + n];
    }
    y += Dv * xv;
    int li;
    if (k == 0)      li = l;
    else if (k == 1) { int w = l / Hh, hh2 = l % Hh; li = hh2 * Ww + w; }
    else if (k == 2) li = Ll - 1 - l;
    else             { int l2 = Ll - 1 - l; int w = l2 / Hh, hh2 = l2 % Hh; li = hh2 * Ww + w; }
    yk[((long)b * Ll + li) * Dd + d] = y;
  }
}

// Sum 4 directions, LayerNorm over D, gate with silu(z). One wave per row.
__global__ void merge_ln_gate(const float* __restrict__ ys, const float* __restrict__ xz,
                              const float* __restrict__ onw, const float* __restrict__ onb,
                              float* __restrict__ gated) {
  const int r = blockIdx.x;
  const int lane = threadIdx.x;  // 32
  const long kS = (long)ROWS * Dd;
  float v[6];
  float sum = 0.f;
#pragma unroll
  for (int j = 0; j < 6; ++j) {
    const int d = lane + j * 32;
    const long off = (long)r * Dd + d;
    float s = ys[off] + ys[kS + off] + ys[2 * kS + off] + ys[3 * kS + off];
    v[j] = s;
    sum += s;
  }
#pragma unroll
  for (int o = 16; o > 0; o >>= 1) sum += __shfl_xor(sum, o, 32);
  const float mu = sum / (float)Dd;
  float var = 0.f;
#pragma unroll
  for (int j = 0; j < 6; ++j) { float dd = v[j] - mu; var += dd * dd; }
#pragma unroll
  for (int o = 16; o > 0; o >>= 1) var += __shfl_xor(var, o, 32);
  const float rs = rsqrtf(var / (float)Dd + 1e-6f);
#pragma unroll
  for (int j = 0; j < 6; ++j) {
    const int d = lane + j * 32;
    const float zn = xz[(long)r * (2 * Dd) + Dd + d];
    const float sil = zn / (1.f + __expf(-zn));
    gated[(long)r * Dd + d] = ((v[j] - mu) * rs * onw[d] + onb[d]) * sil;
  }
}

__global__ void pool_tokens(const float* __restrict__ tn, float* __restrict__ pooled) {
  int idx = blockIdx.x * blockDim.x + threadIdx.x;
  if (idx >= Bsz * Cc) return;
  int b = idx / Cc, c = idx % Cc;
  float s = 0.f;
  for (int l = 0; l < Ll; ++l) s += tn[((long)b * Ll + l) * Cc + c];
  pooled[idx] = s / (float)Ll;
}

} // namespace

extern "C" void kernel_launch(void* const* d_in, const int* in_sizes, int n_in,
                              void* d_out, int out_size, void* d_ws, size_t ws_size,
                              hipStream_t stream) {
  (void)in_sizes; (void)n_in; (void)out_size; (void)ws_size;
  const float* x         = (const float*)d_in[0];
  const float* patch_w   = (const float*)d_in[1];
  const float* patch_b   = (const float*)d_in[2];
  const float* pos       = (const float*)d_in[3];
  const float* ln1_w     = (const float*)d_in[4];
  const float* ln1_b     = (const float*)d_in[5];
  const float* in_proj_w = (const float*)d_in[6];
  const float* conv_w    = (const float*)d_in[7];
  const float* conv_b    = (const float*)d_in[8];
  const float* x_proj_w  = (const float*)d_in[9];
  const float* dt_proj_w = (const float*)d_in[10];
  const float* dt_proj_b = (const float*)d_in[11];
  const float* A_logs    = (const float*)d_in[12];
  const float* Ds        = (const float*)d_in[13];
  const float* onw       = (const float*)d_in[14];
  const float* onb       = (const float*)d_in[15];
  const float* out_proj_w= (const float*)d_in[16];
  const float* norm_w    = (const float*)d_in[17];
  const float* norm_b    = (const float*)d_in[18];
  const float* head_w    = (const float*)d_in[19];
  const float* head_b    = (const float*)d_in[20];
  float* out = (float*)d_out;

  // ---- f32 workspace ----
  float* ws = (float*)d_ws;
  size_t o = 0;
  float* t      = ws + o; o += (size_t)ROWS * Cc;
  float* tn     = ws + o; o += (size_t)ROWS * Cc;
  float* xz     = ws + o; o += (size_t)ROWS * 2 * Dd;
  float* xst    = ws + o; o += (size_t)Kk * ROWS * Dd;
  float* bcb    = ws + o; o += (size_t)Kk * ROWS * BCW;
  float* dts    = ws + o; o += (size_t)Kk * ROWS * Dd;
  float* ys     = ws + o; o += (size_t)Kk * ROWS * Dd;
  float* gated  = ws + o; o += (size_t)ROWS * Dd;
  float* pooled = ws + o; o += (size_t)Bsz * Cc;
  float* col = dts;  // im2col scratch aliased onto dts (disjoint lifetimes)

  // ---- f16 packed-weight pool (256B aligned after the f32 region) ----
  size_t hoff = (o * sizeof(float) + 255) & ~(size_t)255;
  _Float16* hp = (_Float16*)((char*)d_ws + hoff);
  size_t ho = 0;
  _Float16* wp_patch = hp + ho; ho += (size_t)Cc * 768;
  _Float16* wp_in    = hp + ho; ho += (size_t)DEPTH * 2 * Dd * Cc;
  _Float16* wp_bc    = hp + ho; ho += (size_t)DEPTH * Kk * BCW * Dd;
  _Float16* wp_dtf   = hp + ho; ho += (size_t)DEPTH * Kk * Dd * Dd;
  _Float16* wp_out   = hp + ho; ho += (size_t)DEPTH * Cc * Dd;
  _Float16* wp_head  = hp + ho; ho += (size_t)NCLS * Cc;

  const dim3 blk256(256);
  auto nb = [](int total) { return dim3((total + 255) / 256); };

  // ---- per-call weight prep (independent, all up front) ----
  prep_patch  <<<nb(Cc * 768), blk256, 0, stream>>>(patch_w, wp_patch);
  prep_inproj <<<nb(DEPTH * 2 * Dd * Cc), blk256, 0, stream>>>(in_proj_w, wp_in);
  prep_bc     <<<nb(DEPTH * Kk * BCW * Dd), blk256, 0, stream>>>(x_proj_w, wp_bc);
  prep_dtfused<<<nb(DEPTH * Kk * Dd * Dd), blk256, 0, stream>>>(x_proj_w, dt_proj_w, wp_dtf);
  prep_outproj<<<nb(DEPTH * Cc * Dd), blk256, 0, stream>>>(out_proj_w, wp_out);
  prep_head   <<<nb(NCLS * Cc), blk256, 0, stream>>>(head_w, wp_head);

  // ---- patch embed ----
  im2col_patch<<<nb(ROWS * 768), blk256, 0, stream>>>(x, col);
  gemm_wmma<<<dim3(ROWS / 32, Cc / 32, 1), dim3(32), 0, stream>>>(
      col, 0, wp_patch, 0, t, 0, ROWS, Cc, 768, 768, Cc, 0, nullptr, 0);
  patch_epilogue<<<nb(ROWS * Cc), blk256, 0, stream>>>(t, patch_b, pos);

  // ---- 12 SS2D layers ----
  for (int i = 0; i < DEPTH; ++i) {
    layernorm_rows<<<dim3((ROWS + 127) / 128), dim3(128), 0, stream>>>(
        t, tn, ln1_w + (size_t)i * Cc, ln1_b + (size_t)i * Cc, ROWS, Cc);

    gemm_wmma<<<dim3(ROWS / 32, (2 * Dd) / 32, 1), dim3(32), 0, stream>>>(
        tn, 0, wp_in + (size_t)i * 2 * Dd * Cc, 0, xz, 0,
        ROWS, 2 * Dd, Cc, Cc, 2 * Dd, 0, nullptr, 0);

    dwconv_scan_prep<<<nb(ROWS * Dd), blk256, 0, stream>>>(
        xz, conv_w + (size_t)i * Dd * 9, conv_b + (size_t)i * Dd, xst);

    // B|C projection (N = 32 exactly), batched over 4 directions
    gemm_wmma<<<dim3(ROWS / 32, 1, Kk), dim3(32), 0, stream>>>(
        xst, (long)ROWS * Dd, wp_bc + (size_t)i * Kk * BCW * Dd, (long)BCW * Dd,
        bcb, (long)ROWS * BCW, ROWS, BCW, Dd, Dd, BCW, 0, nullptr, 0);

    // fused dt = xs . (x_proj o dt_proj); epilogue: softplus(acc + dt_bias)
    gemm_wmma<<<dim3(ROWS / 32, Dd / 32, Kk), dim3(32), 0, stream>>>(
        xst, (long)ROWS * Dd, wp_dtf + (size_t)i * Kk * Dd * Dd, (long)Dd * Dd,
        dts, (long)ROWS * Dd, ROWS, Dd, Dd, Dd, Dd,
        1, dt_proj_b + (size_t)i * Kk * Dd, (long)Dd);

    selective_scan<<<dim3((Kk * Bsz * Dd + 127) / 128), dim3(128), 0, stream>>>(
        xst, dts, bcb,
        A_logs + (size_t)i * Kk * Dd * Nn, Ds + (size_t)i * Kk * Dd, ys);

    merge_ln_gate<<<dim3(ROWS), dim3(32), 0, stream>>>(
        ys, xz, onw + (size_t)i * Dd, onb + (size_t)i * Dd, gated);

    // out_proj accumulates directly into the residual stream t (mode 2)
    gemm_wmma<<<dim3(ROWS / 32, Cc / 32, 1), dim3(32), 0, stream>>>(
        gated, 0, wp_out + (size_t)i * Cc * Dd, 0, t, 0,
        ROWS, Cc, Dd, Dd, Cc, 2, nullptr, 0);
  }

  // ---- final LN + mean pool + head (bias fused, mode 3) ----
  layernorm_rows<<<dim3((ROWS + 127) / 128), dim3(128), 0, stream>>>(
      t, tn, norm_w, norm_b, ROWS, Cc);
  pool_tokens<<<nb(Bsz * Cc), blk256, 0, stream>>>(tn, pooled);
  gemm_wmma<<<dim3(1, 2, 1), dim3(32), 0, stream>>>(
      pooled, 0, wp_head, 0, out, 0, Bsz, NCLS, Cc, Cc, NCLS, 3, head_b, 0);
}
